// SoftmaxLUT_66288525246508
// MI455X (gfx1250) — compile-verified, matrix operand order checked by reading, not run on hardware
//
#include <hip/hip_runtime.h>
#include <hip/hip_bf16.h>

typedef __attribute__((ext_vector_type(8))) int v8i;

__device__ __forceinline__ int clampi(int v, int lo, int hi) {
    return v < lo ? lo : (v > hi ? hi : v);
}

// One wave32 processes a 16-row x 64-col tile.
//   lane L (L<16)  : row M=L,    K in {0..7, 16..23, 32..39, 48..55}
//   lane L (L>=16) : row M=L-16, K in {8..15, 24..31, 40..47, 56..63}
// This matches the V_WMMA_I32_16X16X64_IU8 A-matrix M<->lane mapping; the
// denominator row sum is computed by one WMMA against an all-ones u8 B matrix.
__global__ __launch_bounds__(256) void softmax_lut_wmma_kernel(
    const int*   __restrict__ qin,      // [rows, 64] int32
    const int*   __restrict__ num_lut,  // [256]
    const int*   __restrict__ den_lut,  // [256]
    const float* __restrict__ s_out_p,  // scalar
    const int*   __restrict__ z_x_p,    // scalar
    const int*   __restrict__ tzp_p,    // scalar
    const int*   __restrict__ z_y_p,    // scalar
    float*       __restrict__ out_q,    // [rows*64] q_out as float
    float*       __restrict__ out_dq,   // [rows*64] dequantized
    long long rows)
{
    __shared__ unsigned lut[256];       // [23:16]=den byte, [15:0]=num u16

    const int tid   = (int)threadIdx.x;
    const int z_x   = z_x_p[0];
    const int tzp   = tzp_p[0];
    const int z_y   = z_y_p[0];
    const float s_o = s_out_p[0];

    // Build fused LUT: one ds_load_b32 per element later instead of 2 gathers.
    {
        int nv = clampi(num_lut[tid], 0, 0xFFFF);        // num fits u16 (<=32385)
        int dv = clampi(den_lut[tid] - tzp, 0, 0xFF);    // den-zp fits u8 (<=127)
        lut[tid] = (unsigned)nv | ((unsigned)dv << 16);
    }
    __syncthreads();

    const int lane = tid & 31;
    const int wave = tid >> 5;
    const int mrow = lane & 15;
    const int khi  = (lane & 16) ? 8 : 0;

    const long long tileRow = (long long)blockIdx.x * 128 + (long long)wave * 16;
    if (tileRow >= rows) return;                          // wave-uniform exit

    const long long row = tileRow + mrow;
    const int* rp = qin + row * 64;

    // ---- load this lane's 32 elements (4 chunks of 8 contiguous ints) ----
    int arr[32];
    #pragma unroll
    for (int j = 0; j < 4; ++j) {
        const int ks = j * 16 + khi;
        int4 lo = *(const int4*)(rp + ks);
        int4 hi = *(const int4*)(rp + ks + 4);
        arr[j*8+0]=lo.x; arr[j*8+1]=lo.y; arr[j*8+2]=lo.z; arr[j*8+3]=lo.w;
        arr[j*8+4]=hi.x; arr[j*8+5]=hi.y; arr[j*8+6]=hi.z; arr[j*8+7]=hi.w;
    }

    // ---- row max: 32 local values + one xor-16 shuffle (lane pair L,L+16) ----
    int mx = arr[0];
    #pragma unroll
    for (int i = 1; i < 32; ++i) mx = arr[i] > mx ? arr[i] : mx;
    mx = max(mx, __shfl_xor(mx, 16, 32));

    // ---- gather fused LUT; keep numerators, pack den bytes into A-matrix ----
    v8i A;
    #pragma unroll
    for (int j = 0; j < 4; ++j) {
        unsigned p0 = 0, p1 = 0;
        #pragma unroll
        for (int e = 0; e < 8; ++e) {
            const int i = j*8 + e;
            int a = clampi(arr[i] - mx, -256, 255);       // intsub @ 9 bits
            a = clampi(a + z_x, -256, 255);               // intadd @ 9 bits
            const int idx = clampi(a + 128, 0, 255);      // a - MIN_Q
            const unsigned entry = lut[idx];              // ds_load_b32
            arr[i] = (int)(entry & 0xFFFFu);              // numerator
            const unsigned d = entry >> 16;               // denominator byte
            if (e < 4) p0 |= d << (8*e);
            else       p1 |= d << (8*(e-4));
        }
        A[2*j]   = (int)p0;
        A[2*j+1] = (int)p1;
    }

    // ---- denominator row-sum via WMMA IU8 against all-ones B ----
    v8i B;
    #pragma unroll
    for (int k = 0; k < 8; ++k) B[k] = 0x01010101;
    v8i C = {0,0,0,0,0,0,0,0};
    v8i S = __builtin_amdgcn_wmma_i32_16x16x64_iu8(false, A, false, B, C,
                                                   false, false);

    // C/D layout: S[r] on lane n = rowsum(M = r + 8*(n>=16)) (cols identical).
    // Lane n locally selects S[n&7]; lane L then fetches its row sum from
    // src = (L&7) | ((L&8)<<1) via ds_bpermute.
    const int sel = lane & 7;
    int w;
    {
        const bool b1 = sel & 1, b2 = sel & 2, b4 = sel & 4;
        const int w0 = b4 ? S[4] : S[0];
        const int w1 = b4 ? S[5] : S[1];
        const int w2 = b4 ? S[6] : S[2];
        const int w3 = b4 ? S[7] : S[3];
        const int wa = b2 ? w2 : w0;
        const int wb = b2 ? w3 : w1;
        w = b1 ? wb : wa;
    }
    const int srcLane = (lane & 7) | ((lane & 8) << 1);
    int rs = __shfl(w, srcLane, 32);
    rs = rs == 0 ? 1 : rs;

    // ---- exact truncated division: raw v_rcp_f32 + branchless +-1 fixup ----
    // Fixup makes the result exact, so the 1-ulp rcp is sufficient (quotient
    // <= 255 => absolute error ~6e-5 << 1).
    const float rinv = __builtin_amdgcn_rcpf((float)rs);
    float* oq = out_q  + row * 64;
    float* od = out_dq + row * 64;
    #pragma unroll
    for (int j = 0; j < 4; ++j) {
        const int ks = j * 16 + khi;
        float qo[8], dq[8];
        #pragma unroll
        for (int e = 0; e < 8; ++e) {
            const int n = arr[j*8+e];                      // numerator >= 0
            int t = (int)((float)n * rinv);
            const int prod = t * rs;
            t -= (int)(prod > n);                          // branchless exact
            t += (int)(prod + rs <= n);                    // trunc fixup
            t = clampi(t, -256, 255);
            const int qv = clampi(t + z_y, -128, 127);
            qo[e] = (float)qv;
            dq[e] = (float)(qv - z_y) * s_o;
        }
        *(float4*)(oq + ks)     = make_float4(qo[0],qo[1],qo[2],qo[3]);
        *(float4*)(oq + ks + 4) = make_float4(qo[4],qo[5],qo[6],qo[7]);
        *(float4*)(od + ks)     = make_float4(dq[0],dq[1],dq[2],dq[3]);
        *(float4*)(od + ks + 4) = make_float4(dq[4],dq[5],dq[6],dq[7]);
    }
}

extern "C" void kernel_launch(void* const* d_in, const int* in_sizes, int n_in,
                              void* d_out, int out_size, void* d_ws, size_t ws_size,
                              hipStream_t stream) {
    const int*   qin     = (const int*)d_in[0];
    const int*   num_lut = (const int*)d_in[1];
    const int*   den_lut = (const int*)d_in[2];
    const float* s_out   = (const float*)d_in[3];
    const int*   z_x     = (const int*)d_in[4];
    const int*   tzp     = (const int*)d_in[5];
    const int*   z_y     = (const int*)d_in[6];

    const long long N    = (long long)in_sizes[0];   // 512*8*64*64
    const long long rows = N / 64;
    float* out_q  = (float*)d_out;                   // q_out as float
    float* out_dq = out_q + N;                       // dq_out

    const int blocks = (int)((rows + 127) / 128);    // 8 waves/block, 16 rows/wave
    softmax_lut_wmma_kernel<<<blocks, 256, 0, stream>>>(
        qin, num_lut, den_lut, s_out, z_x, tzp, z_y, out_q, out_dq, rows);
}
